// DeformableCrossAttention_7164005449748
// MI455X (gfx1250) — matrix operand order repaired and checked by reading, not compiled, not run on hardware
//
#include <hip/hip_runtime.h>
#include <math.h>

typedef __attribute__((ext_vector_type(16))) __bf16 v16bf;
typedef __attribute__((ext_vector_type(8)))  __bf16 v8bf;
typedef __attribute__((ext_vector_type(8)))  float  v8f;

#define BS_  32
#define NTOK 4096
#define E_   1024
#define C_   64
#define P_   16
#define H_   16
#define NT_  64      // E/16 column tiles
#define KB_  32      // E/32 k-steps per 16x16x32 WMMA chain

// -------------------------------------------------------------------------
// Pack a (E x E) row-major fp32 weight matrix into bf16 WMMA B-fragment
// order:  packed[((kb*NT + nt)*32 + lane)*16 + e]
//   lane <  16 : N = nt*16 + lane,        K = kb*32 + {0..7,16..23}[e]
//   lane >= 16 : N = nt*16 + (lane-16),   K = kb*32 + {8..15,24..31}[e]
// -------------------------------------------------------------------------
__global__ void __launch_bounds__(256)
k_pack(const float* __restrict__ W, __bf16* __restrict__ packed) {
  unsigned t = blockIdx.x * 256u + threadIdx.x;     // 1M elements total
  int e    = t & 15;
  int lane = (t >> 4) & 31;
  int nt   = (t >> 9) & 63;
  int kb   = t >> 15;
  int K    = kb * 32 + (lane >> 4) * 8 + (e & 7) + (e >> 3) * 16;
  int Nc   = nt * 16 + (lane & 15);
  packed[t] = (__bf16)W[(size_t)K * E_ + Nc];
}

// -------------------------------------------------------------------------
// Sampling indices + point weights.  One wave per (b,c).
// -------------------------------------------------------------------------
__global__ void __launch_bounds__(32)
k_sample(const float* __restrict__ query,
         const float* __restrict__ W_pts, const float* __restrict__ b_pts,
         const float* __restrict__ W_w,   const float* __restrict__ b_w,
         int* __restrict__ idx, float* __restrict__ wgt) {
  int bc   = blockIdx.x;
  int lane = threadIdx.x;
  int half = lane >> 4;
  int p    = lane & 15;
  const float* q = query + (size_t)bc * E_;

  float ap = 0.f, aw = 0.f;
  int e0 = half * 512;
  for (int e = e0; e < e0 + 512; ++e) {
    float qe = q[e];
    ap = fmaf(qe, W_pts[e * 16 + p], ap);
    aw = fmaf(qe, W_w  [e * 16 + p], aw);
  }
  ap += __shfl_xor(ap, 16);
  aw += __shfl_xor(aw, 16);
  ap += b_pts[p];
  aw += b_w[p];

  float sg = 1.f / (1.f + __expf(-ap));
  int id = (int)floorf(sg * (float)NTOK);
  id = id < 0 ? 0 : (id > NTOK - 1 ? NTOK - 1 : id);

  float mx = aw;
  mx = fmaxf(mx, __shfl_xor(mx, 1)); mx = fmaxf(mx, __shfl_xor(mx, 2));
  mx = fmaxf(mx, __shfl_xor(mx, 4)); mx = fmaxf(mx, __shfl_xor(mx, 8));
  float ex = __expf(aw - mx);
  float s  = ex;
  s += __shfl_xor(s, 1); s += __shfl_xor(s, 2);
  s += __shfl_xor(s, 4); s += __shfl_xor(s, 8);

  if (half == 0) {
    idx[bc * 16 + p] = id;
    wgt[bc * 16 + p] = ex / s;
  }
}

// -------------------------------------------------------------------------
// Fused: gather+scale -> LDS(bf16); K-projection via WMMA with q.k energy
// accumulated in registers per head (one butterfly per head, no atomics);
// softmax over P per head; V-projection via WMMA contracted with attention
// weights.  K and V are never materialized.  One 8-wave block per (b,c);
// wave w owns n-tiles [8w, 8w+8) == heads 2w and 2w+1.
// -------------------------------------------------------------------------
__global__ void __launch_bounds__(256)
k_attn(const float* __restrict__ input, const float* __restrict__ query,
       const int* __restrict__ idx, const float* __restrict__ wgt,
       const __bf16* __restrict__ pWk, const __bf16* __restrict__ pWv,
       const float* __restrict__ b_k, const float* __restrict__ b_v,
       __bf16* __restrict__ attn_out) {
  __shared__ __align__(32) __bf16 Abuf[P_ * E_];   // 32 KB bf16 deform tile
  __shared__ float energy[H_ * P_];                // 1 KB
  __shared__ float att[H_ * P_];                   // 1 KB
  __shared__ int   s_idx[P_];
  __shared__ float s_w[P_];

  int bc = blockIdx.x;
  int b  = bc >> 6;            // C_ == 64
  int tid  = threadIdx.x;
  int wave = tid >> 5;
  int lane = tid & 31;
  int half = lane >> 4;
  int l16  = lane & 15;

  if (tid < P_) { s_idx[tid] = idx[bc * P_ + tid]; s_w[tid] = wgt[bc * P_ + tid]; }
  __syncthreads();

  // ---- gather + scale + fp32->bf16, 16 rows of 1024 into LDS ----
  for (int t = tid; t < P_ * (E_ / 4); t += 256) {
    int p  = t >> 8;                 // E_/4 = 256 float4 chunks per row
    int ch = t & 255;
    const float4 g =
        ((const float4*)(input + ((size_t)b * NTOK + s_idx[p]) * E_))[ch];
    float wp = s_w[p];
    __bf16* d = &Abuf[p * E_ + ch * 4];
    d[0] = (__bf16)(g.x * wp); d[1] = (__bf16)(g.y * wp);
    d[2] = (__bf16)(g.z * wp); d[3] = (__bf16)(g.w * wp);
  }
  __syncthreads();

  // One 16x16 output tile over K=1024 via 32 chained bf16 WMMAs.
  // A-fragment (16x32 bf16): lane<16 -> M=l16, K = k0+{0..7,16..23};
  //                          lane>=16 -> M=l16, K = k0+{8..15,24..31}.
  auto mma_tile = [&](const __bf16* __restrict__ packed, int nt) -> v8f {
    v8f acc = {};
    const __bf16* bp = packed + ((size_t)nt * 32 + lane) * 16;
    #pragma unroll 4
    for (int kb = 0; kb < KB_; ++kb) {
      int aoff = kb * 32 + half * 8;
      union { v16bf v; v8bf h[2]; } A, B;
      A.h[0] = *(const v8bf*)&Abuf[l16 * E_ + aoff];
      A.h[1] = *(const v8bf*)&Abuf[l16 * E_ + aoff + 16];
      B.v    = *(const v16bf*)bp;
      __builtin_prefetch(bp + (size_t)NT_ * 32 * 16, 0, 3);
      acc = __builtin_amdgcn_wmma_f32_16x16x32_bf16(
          false, A.v, false, B.v, (short)0, acc, false, false);
      bp += (size_t)NT_ * 32 * 16;
    }
    return acc;
  };

  const float* qrow = query + (size_t)bc * E_;

  // ---- K phase: per head, accumulate q-scaled column partials across the
  //      head's 4 n-tiles in registers; one butterfly + one LDS store per
  //      (head, row) -- no atomics, no per-tile reduction chains.
  for (int hh = 0; hh < 2; ++hh) {
    int h = wave * 2 + hh;
    float tp[8];
    #pragma unroll
    for (int v = 0; v < 8; ++v) tp[v] = 0.f;

    for (int j = 0; j < 4; ++j) {
      int nt = h * 4 + j;
      v8f acc = mma_tile(pWk, nt);
      int n0 = nt * 16;
      float qv = qrow[n0 + l16];
      float bk = b_k[n0 + l16];
      #pragma unroll
      for (int v = 0; v < 8; ++v)
        tp[v] = fmaf(acc[v] + bk, qv, tp[v]);   // col N = n0+l16, row M = half*8+v
    }
    #pragma unroll
    for (int v = 0; v < 8; ++v) {
      float t = tp[v];
      t += __shfl_xor(t, 1); t += __shfl_xor(t, 2);
      t += __shfl_xor(t, 4); t += __shfl_xor(t, 8);
      if (l16 == v) energy[h * 16 + half * 8 + v] = t;
    }
  }
  __syncthreads();

  // ---- softmax over P per head (scaled by 1/sqrt(E)) ----
  if (tid < H_) {
    const float scale = 0.03125f;     // 1/sqrt(1024)
    int h = tid;
    float mx = -3.0e38f;
    for (int p = 0; p < P_; ++p) mx = fmaxf(mx, energy[h * 16 + p] * scale);
    float ev[P_]; float s = 0.f;
    for (int p = 0; p < P_; ++p) {
      ev[p] = __expf(energy[h * 16 + p] * scale - mx);
      s += ev[p];
    }
    float inv = 1.f / s;
    for (int p = 0; p < P_; ++p) att[h * 16 + p] = ev[p] * inv;
  }
  __syncthreads();

  // ---- V phase: out[n] = sum_p att[h][p] * (A@Wv + bv)[p][n] ----
  for (int nt = wave * 8; nt < wave * 8 + 8; ++nt) {
    v8f acc = mma_tile(pWv, nt);
    int n0 = nt * 16;
    int h  = n0 >> 6;
    float bv = b_v[n0 + l16];
    const float* atth = &att[h * 16 + half * 8];   // rows M = half*8 + v
    float s = 0.f;
    #pragma unroll
    for (int v = 0; v < 8; ++v) s = fmaf(acc[v] + bv, atth[v], s);
    s += __shfl_xor(s, 16);          // combine rows 0-7 with rows 8-15
    if (half == 0)
      attn_out[(size_t)bc * E_ + n0 + l16] = (__bf16)s;
  }
}

// -------------------------------------------------------------------------
// Output projection: (2048 x 1024) bf16 @ packed W_o (bf16) + b_o -> fp32.
// One 16x16 tile per wave, 8 waves per block.
// -------------------------------------------------------------------------
__global__ void __launch_bounds__(256)
k_proj(const __bf16* __restrict__ A, const __bf16* __restrict__ pWo,
       const float* __restrict__ b_o, float* __restrict__ out) {
  int job  = blockIdx.x * 8 + (threadIdx.x >> 5);   // 8192 tile jobs
  int mt   = job >> 6;
  int nt   = job & 63;
  int lane = threadIdx.x & 31;
  int half = lane >> 4;
  int l16  = lane & 15;

  v8f acc = {};
  const __bf16* bp = pWo + ((size_t)nt * 32 + lane) * 16;
  const __bf16* ap = A + (size_t)(mt * 16 + l16) * E_ + half * 8;
  #pragma unroll 4
  for (int kb = 0; kb < KB_; ++kb) {
    union { v16bf v; v8bf h[2]; } Af, Bf;
    Af.h[0] = *(const v8bf*)(ap);
    Af.h[1] = *(const v8bf*)(ap + 16);
    Bf.v    = *(const v16bf*)bp;
    __builtin_prefetch(bp + (size_t)NT_ * 32 * 16, 0, 3);
    acc = __builtin_amdgcn_wmma_f32_16x16x32_bf16(
        false, Af.v, false, Bf.v, (short)0, acc, false, false);
    ap += 32;
    bp += (size_t)NT_ * 32 * 16;
  }
  int n = nt * 16 + l16;
  float bo = b_o[n];
  #pragma unroll
  for (int v = 0; v < 8; ++v) {
    int row = mt * 16 + half * 8 + v;
    out[(size_t)row * E_ + n] = acc[v] + bo;
  }
}

// -------------------------------------------------------------------------
extern "C" void kernel_launch(void* const* d_in, const int* in_sizes, int n_in,
                              void* d_out, int out_size, void* d_ws, size_t ws_size,
                              hipStream_t stream) {
  const float* input = (const float*)d_in[0];
  const float* query = (const float*)d_in[1];
  const float* W_pts = (const float*)d_in[2];
  const float* b_pts = (const float*)d_in[3];
  const float* W_w   = (const float*)d_in[4];
  const float* b_w   = (const float*)d_in[5];
  const float* W_k   = (const float*)d_in[6];
  const float* b_k   = (const float*)d_in[7];
  const float* W_v   = (const float*)d_in[8];
  const float* b_v   = (const float*)d_in[9];
  const float* W_o   = (const float*)d_in[10];
  const float* b_o   = (const float*)d_in[11];
  float* out = (float*)d_out;

  // workspace layout (~10.3 MB)
  char* w = (char*)d_ws;
  int*    idxbuf = (int*)   (w);                               // 128 KB
  float*  wgtbuf = (float*) (w + (128 << 10));                 // 128 KB
  __bf16* pWk    = (__bf16*)(w + (256 << 10));                 // 2 MB
  __bf16* pWv    = (__bf16*)(w + (256 << 10) + (2u << 20));    // 2 MB
  __bf16* pWo    = (__bf16*)(w + (256 << 10) + (4u << 20));    // 2 MB
  __bf16* attn   = (__bf16*)(w + (256 << 10) + (6u << 20));    // 4 MB

  k_pack  <<<4096, 256, 0, stream>>>(W_k, pWk);
  k_pack  <<<4096, 256, 0, stream>>>(W_v, pWv);
  k_pack  <<<4096, 256, 0, stream>>>(W_o, pWo);
  k_sample<<<BS_ * C_, 32, 0, stream>>>(query, W_pts, b_pts, W_w, b_w,
                                        idxbuf, wgtbuf);
  k_attn  <<<BS_ * C_, 256, 0, stream>>>(input, query, idxbuf, wgtbuf,
                                         pWk, pWv, b_k, b_v, attn);
  k_proj  <<<1024, 256, 0, stream>>>(attn, pWo, b_o, out);
}
